// VGG16_3DNetwork_4080218931708
// MI455X (gfx1250) — matrix operand order repaired and checked by reading, not compiled, not run on hardware
//
#include <hip/hip_runtime.h>

typedef __attribute__((ext_vector_type(16))) __bf16 v16bf;
typedef __attribute__((ext_vector_type(8)))  float  v8f;

// Activation column permutation (per 32-column chunk) so that the WMMA A-operand
// per-lane fragment is one contiguous 32B load:
//   stored order = [K0..7, K16..23, K8..15, K24..31]
// Involution: perm(perm(c)) == c.
__device__ __forceinline__ int permc(int c)
{
    int r = c & 31, b = c & ~31;
    int g = (r >> 3) & 3;               // octet index 0..3
    r += (g == 1) ? 8 : ((g == 2) ? -8 : 0);   // swap octets 1 <-> 2
    return b + r;
}

// ---------------------------------------------------------------------------
// Sparse conv as gather + WMMA:  y[i] = sum_k x_pad[im[k][i]] @ W[k]
// One wave computes a 16-row x (16*CT)-col tile of y with CT accumulators.
// act : (N+1) x cin bf16, permuted columns, row N is all zeros (pad row)
// im  : 27 x N   int32 (pad index == N)
// wp  : 27 x cout x cin bf16 (permuted cin order, same as act -> matches A K map)
// y   : N x cout f32 (logical order)
// ---------------------------------------------------------------------------
template <int CT>
__global__ __launch_bounds__(32) void conv_wmma_k(
    const __bf16* __restrict__ act, const int* __restrict__ im,
    const __bf16* __restrict__ wp, float* __restrict__ y,
    int N, int cin, int cout)
{
    const int lane = threadIdx.x;
    const int m    = lane & 15;   // row (A) / col (B,D) within tile
    const int half = lane >> 4;   // K-chunk selector
    const int r0   = blockIdx.x * 16;
    const int c0   = blockIdx.y * (16 * CT);

    v8f acc[CT];
#pragma unroll
    for (int t = 0; t < CT; ++t) acc[t] = (v8f){};

    const int  ri    = r0 + m;
    const bool rowok = ri < N;

    for (int k = 0; k < 27; ++k) {
        const int src = rowok ? im[k * N + ri] : N;   // N -> zero pad row
        const __bf16* xrow = act + (long)src * cin + half * 16;
        // B base for this lane: column (c0+m), K offset half*16 within row
        const __bf16* w0 = wp + ((long)k * cout + (c0 + m)) * cin + half * 16;

        for (int ci = 0; ci < cin; ci += 32) {
            // A: one contiguous 32B load thanks to permuted activation storage
            v16bf a = *(const v16bf*)(xrow + ci);
#pragma unroll
            for (int t = 0; t < CT; ++t) {
                v16bf b = *(const v16bf*)(w0 + (long)(16 * t) * cin + ci);
                acc[t] = __builtin_amdgcn_wmma_f32_16x16x32_bf16(
                    false, a, false, b, (short)0, acc[t], false, false);
            }
        }
    }

    // D layout: lane holds column m; VGPR j holds row half*8 + j
#pragma unroll
    for (int j = 0; j < 8; ++j) {
        const int row = r0 + half * 8 + j;
        if (row < N) {
            float* yr = y + (long)row * cout + c0 + m;
#pragma unroll
            for (int t = 0; t < CT; ++t) yr[16 * t] = acc[t][j];
        }
    }
}

// ---- pack W[27][cinReal][cout] f32 -> wp[27][cout][cinPad] bf16 (zero-pad) ----
__global__ void pack_w_k(const float* __restrict__ w, __bf16* __restrict__ wp,
                         int cinReal, int cinPad, int cout, long total)
{
    long idx = (long)blockIdx.x * blockDim.x + threadIdx.x;
    if (idx >= total) return;
    int  ci  = (int)(idx % cinPad);
    long r   = idx / cinPad;
    int  co  = (int)(r % cout);
    int  k   = (int)(r / cout);
    float v  = (permc(ci) < cinReal) ? w[((long)k * cinReal + permc(ci)) * cout + co] : 0.0f;
    wp[idx] = (__bf16)v;
}

// ---- x (N x 3 f32) -> act (N+1) x 32 bf16, permuted cols, zero-padded ----
__global__ void load_input_k(const float* __restrict__ x, __bf16* __restrict__ act,
                             int N, long total)
{
    long idx = (long)blockIdx.x * blockDim.x + threadIdx.x;
    if (idx >= total) return;
    int i = (int)(idx >> 5);
    int c = permc((int)(idx & 31));      // logical channel at this stored slot
    float v = (i < N && c < 3) ? x[i * 3 + c] : 0.0f;
    act[idx] = (__bf16)v;
}

__global__ void zero_f32_k(float* __restrict__ p, long n)
{
    long i = (long)blockIdx.x * blockDim.x + threadIdx.x;
    if (i < n) p[i] = 0.0f;
}

// ---- BN pass 1: per-channel sum / sumsq on logical-order f32 y ----
__global__ void bn_stats_k(const float* __restrict__ y, float* __restrict__ stats,
                           int N, int C, int rowsPerBlock)
{
    int r0 = blockIdx.x * rowsPerBlock;
    int r1 = r0 + rowsPerBlock; if (r1 > N) r1 = N;
    for (int c = threadIdx.x; c < C; c += blockDim.x) {
        float s = 0.f, q = 0.f;
        for (int i = r0; i < r1; ++i) {
            float v = y[(long)i * C + c];
            s += v; q += v * v;
        }
        atomicAdd(&stats[c], s);
        atomicAdd(&stats[C + c], q);
    }
}

// ---- BN pass 2 + ReLU -> bf16 act (permuted cols; zero pad row i==N) ----
__global__ void bn_apply_k(const float* __restrict__ y, const float* __restrict__ stats,
                           const float* __restrict__ gamma, const float* __restrict__ beta,
                           __bf16* __restrict__ act, int N, int C, long total)
{
    long idx = (long)blockIdx.x * blockDim.x + threadIdx.x;
    if (idx >= total) return;
    int i = (int)(idx / C);
    if (i == N) { act[idx] = (__bf16)0.0f; return; }
    int c = permc((int)(idx % C));       // logical channel at this stored slot
    float inv  = 1.0f / (float)N;
    float mval = stats[c] * inv;
    float var  = stats[C + c] * inv - mval * mval;
    float sc   = rsqrtf(var + 1e-5f) * gamma[c];
    float v    = (y[(long)i * C + c] - mval) * sc + beta[c];
    act[idx] = (__bf16)fmaxf(v, 0.0f);
}

// ---- segment max pool in stored order (ReLU => >=0 => int-bit atomicMax ok) ----
__global__ void pool_max_k(const __bf16* __restrict__ act, const int* __restrict__ seg,
                           float* __restrict__ pool, int N, int C, long total)
{
    long idx = (long)blockIdx.x * blockDim.x + threadIdx.x;
    if (idx >= total) return;
    int i = (int)(idx / C), p = (int)(idx % C);
    float v = (float)act[idx];
    atomicMax((int*)&pool[(long)seg[i] * C + p], __float_as_int(v));
}

// ---- pooled f32 (stored order) -> bf16 act (stored order) with zero pad row ----
__global__ void f32_to_act_k(const float* __restrict__ src, __bf16* __restrict__ act,
                             int N, int C, long total)
{
    long idx = (long)blockIdx.x * blockDim.x + threadIdx.x;
    if (idx >= total) return;
    int i = (int)(idx / C);
    act[idx] = (i == N) ? (__bf16)0.0f : (__bf16)src[idx];
}

// ---- global average pool over rows; un-permute to logical channel order ----
__global__ void gmean_k(const __bf16* __restrict__ act, float* __restrict__ mean,
                        int N, int C)
{
    int c = blockIdx.x * blockDim.x + threadIdx.x;  // logical channel
    if (c >= C) return;
    int p = permc(c);                                // stored slot
    float s = 0.f;
    for (int i = 0; i < N; ++i) s += (float)act[(long)i * C + p];
    mean[c] = s / (float)N;
}

// ---- final linear: out[j] = b[j] + sum_c mean[c] * W[c][j] ----
__global__ void head_k(const float* __restrict__ mean, const float* __restrict__ W,
                       const float* __restrict__ b, float* __restrict__ out,
                       int C, int O)
{
    int j = blockIdx.x * blockDim.x + threadIdx.x;
    if (j >= O) return;
    float s = b[j];
    for (int c = 0; c < C; ++c) s += mean[c] * W[(long)c * O + j];
    out[j] = s;
}

// ===========================================================================
static inline long cdivl(long a, long b) { return (a + b - 1) / b; }

extern "C" void kernel_launch(void* const* d_in, const int* in_sizes, int n_in,
                              void* d_out, int out_size, void* d_ws, size_t ws_size,
                              hipStream_t stream)
{
    (void)n_in; (void)out_size; (void)ws_size;
    // ---- input layout (setup_inputs dict order, lists flattened) ----
    // 0:x1 1:x2 2..14:conv_ws 15..27:gammas 28..40:betas 41:lin_w 42:lin_b
    // 43..47:in_maps 48..52:out_maps 53..56:pool_maps
    const float* xin[2] = { (const float*)d_in[0], (const float*)d_in[1] };
    const float* convw[13]; const float* gam[13]; const float* bet[13];
    for (int i = 0; i < 13; ++i) {
        convw[i] = (const float*)d_in[2 + i];
        gam[i]   = (const float*)d_in[15 + i];
        bet[i]   = (const float*)d_in[28 + i];
    }
    const float* linW = (const float*)d_in[41];
    const float* linB = (const float*)d_in[42];
    const int* imaps[5]; int Ns[5];
    for (int s = 0; s < 5; ++s) { imaps[s] = (const int*)d_in[43 + s]; Ns[s] = in_sizes[43 + s] / 27; }
    const int* pmaps[4];
    for (int s = 0; s < 4; ++s) pmaps[s] = (const int*)d_in[53 + s];
    float* out = (float*)d_out;

    const int stage_convs[5] = {2, 2, 3, 3, 3};
    const int stage_ch[5]    = {64, 128, 256, 512, 512};
    const int stage_cin[5]   = {32, 64, 128, 256, 512};   // input width per stage (L0 padded)

    // ---- workspace carving (256B aligned) ----
    char* ws = (char*)d_ws;
    size_t off = 0;
    auto carve = [&](size_t bytes) -> void* {
        void* p = (void*)(ws + off);
        off += (bytes + 255) & ~(size_t)255;
        return p;
    };
    long maxAct = 0, maxY = 0, maxPool = 0;
    for (int s = 0; s < 5; ++s) {
        long a1 = (long)(Ns[s] + 1) * stage_cin[s];
        long a2 = (long)(Ns[s] + 1) * stage_ch[s];
        if (a1 > maxAct) maxAct = a1;
        if (a2 > maxAct) maxAct = a2;
        long yy = (long)Ns[s] * stage_ch[s];
        if (yy > maxY) maxY = yy;
        if (s < 4) { long pp = (long)Ns[s + 1] * stage_ch[s]; if (pp > maxPool) maxPool = pp; }
    }
    __bf16* actA  = (__bf16*)carve((size_t)maxAct * 2);
    __bf16* actB  = (__bf16*)carve((size_t)maxAct * 2);
    float*  yf    = (float*)carve((size_t)maxY * 4);
    float*  pool  = (float*)carve((size_t)maxPool * 4);
    float*  stats = (float*)carve(2 * 512 * 4);
    float*  mean  = (float*)carve(512 * 4);
    __bf16* wp    = (__bf16*)carve((size_t)27 * 512 * 512 * 2);

    const int TB = 256;

    for (int rep = 0; rep < 2; ++rep) {
        // stage-0 input: f32 N0x3 -> bf16 (N0+1)x32 padded+permuted
        {
            long tot = (long)(Ns[0] + 1) * 32;
            load_input_k<<<(unsigned)cdivl(tot, TB), TB, 0, stream>>>(xin[rep], actA, Ns[0], tot);
        }
        __bf16* cur = actA;
        __bf16* nxt = actB;
        int li = 0, cin = 32;

        for (int s = 0; s < 5; ++s) {
            const int N = Ns[s];
            for (int cidx = 0; cidx < stage_convs[s]; ++cidx) {
                const int cout = stage_ch[s];
                const int cinReal = (li == 0) ? 3 : cin;

                long wtot = (long)27 * cout * cin;
                pack_w_k<<<(unsigned)cdivl(wtot, TB), TB, 0, stream>>>(convw[li], wp, cinReal, cin, cout, wtot);

                if (cout == 64) {
                    dim3 g((unsigned)cdivl(N, 16), 1);
                    conv_wmma_k<4><<<g, 32, 0, stream>>>(cur, imaps[s], wp, yf, N, cin, cout);
                } else {
                    dim3 g((unsigned)cdivl(N, 16), (unsigned)(cout / 128));
                    conv_wmma_k<8><<<g, 32, 0, stream>>>(cur, imaps[s], wp, yf, N, cin, cout);
                }

                zero_f32_k<<<(unsigned)cdivl(2 * cout, TB), TB, 0, stream>>>(stats, 2 * cout);
                const int rowsPerBlock = 256;
                bn_stats_k<<<(unsigned)cdivl(N, rowsPerBlock), TB, 0, stream>>>(yf, stats, N, cout, rowsPerBlock);

                long atot = (long)(N + 1) * cout;
                bn_apply_k<<<(unsigned)cdivl(atot, TB), TB, 0, stream>>>(yf, stats, gam[li], bet[li], nxt, N, cout, atot);

                __bf16* t = cur; cur = nxt; nxt = t;
                cin = cout; ++li;
            }
            if (s < 4) {
                const int Nn = Ns[s + 1];
                zero_f32_k<<<(unsigned)cdivl((long)Nn * cin, TB), TB, 0, stream>>>(pool, (long)Nn * cin);
                long ptot = (long)N * cin;
                pool_max_k<<<(unsigned)cdivl(ptot, TB), TB, 0, stream>>>(cur, pmaps[s], pool, N, cin, ptot);
                long ctot = (long)(Nn + 1) * cin;
                f32_to_act_k<<<(unsigned)cdivl(ctot, TB), TB, 0, stream>>>(pool, nxt, Nn, cin, ctot);
                __bf16* t = cur; cur = nxt; nxt = t;
            }
        }

        gmean_k<<<2, 256, 0, stream>>>(cur, mean, Ns[4], 512);
        head_k<<<1, 256, 0, stream>>>(mean, linW, linB, out + rep * 256, 512, 256);
    }
}
